// INRLoe_44925357916946
// MI455X (gfx1250) — matrix operand order repaired and checked by reading, not compiled, not run on hardware
//
#include <hip/hip_runtime.h>
#include <hip/hip_bf16.h>

#define NB 16        // batch
#define NP 16384     // points per image
#define NH 256       // hidden dim
#define NL 64        // latent dim

typedef __attribute__((ext_vector_type(16))) _Float16 v16h;
typedef __attribute__((ext_vector_type(8)))  _Float16 v8h;
typedef __attribute__((ext_vector_type(8)))  float    v8f;

__constant__ int c_NE[5] = {8, 16, 64, 256, 1024};
__constant__ int c_TK[5] = {4, 4, 32, 32, 256};

// ---------------------------------------------------------------------------
// Gate + abs-top-k per (b, layer). One block per pair; bitonic sort of |g|
// gives the k-th largest magnitude as a threshold, then compact selection.
// ---------------------------------------------------------------------------
__global__ __launch_bounds__(256) void gate_topk_kernel(
    const float* __restrict__ latents,
    const float* Wg0, const float* Wg1, const float* Wg2, const float* Wg3, const float* Wg4,
    const float* bg0, const float* bg1, const float* bg2, const float* bg3, const float* bg4,
    int* __restrict__ selIdx, float* __restrict__ selVal, int* __restrict__ selCnt)
{
    __shared__ float lat[NL];
    __shared__ float gval[1024];
    __shared__ float skey[1024];
    __shared__ int   scnt;

    const int b   = blockIdx.x;
    const int l   = blockIdx.y;
    const int tid = threadIdx.x;
    const int E   = c_NE[l];
    const int k   = c_TK[l];

    const float* Wg = (l == 0) ? Wg0 : (l == 1) ? Wg1 : (l == 2) ? Wg2 : (l == 3) ? Wg3 : Wg4;
    const float* bg = (l == 0) ? bg0 : (l == 1) ? bg1 : (l == 2) ? bg2 : (l == 3) ? bg3 : bg4;

    if (tid < NL) lat[tid] = latents[(b * 5 + l) * NL + tid];
    if (tid == 0) scnt = 0;
    __syncthreads();

    for (int e = tid; e < E; e += 256) {
        float acc = bg[e];
        const float* w = Wg + e * NL;
#pragma unroll 8
        for (int j = 0; j < NL; ++j) acc += lat[j] * w[j];
        gval[e] = acc;
        skey[e] = fabsf(acc);
    }
    __syncthreads();

    // bitonic sort (descending) of skey[0..E); E is a power of two
    for (int ks = 2; ks <= E; ks <<= 1) {
        for (int js = ks >> 1; js > 0; js >>= 1) {
            for (int i = tid; i < E; i += 256) {
                int ij = i ^ js;
                if (ij > i) {
                    float a = skey[i], c = skey[ij];
                    bool desc = ((i & ks) == 0);
                    if (desc ? (a < c) : (a > c)) { skey[i] = c; skey[ij] = a; }
                }
            }
            __syncthreads();
        }
    }
    const float thr = skey[k - 1];
    __syncthreads();

    const int base = (l * NB + b) * 256;
    for (int e = tid; e < E; e += 256) {
        float g = gval[e];
        if (fabsf(g) >= thr) {
            int s = atomicAdd(&scnt, 1);
            if (s < 256) { selIdx[base + s] = e; selVal[base + s] = g; }
        }
    }
    __syncthreads();
    if (tid == 0) selCnt[l * NB + b] = (scnt < 256) ? scnt : 256;
}

// ---------------------------------------------------------------------------
// Per-sample combined bias: cb[b][o] = sum_j val_j * bp[idx_j*do + o]  (f32)
// ---------------------------------------------------------------------------
__global__ __launch_bounds__(256) void cb_combine_kernel(
    const float* __restrict__ bp, const int* __restrict__ selIdx,
    const float* __restrict__ selVal, const int* __restrict__ selCnt,
    float* __restrict__ cb, int dol)
{
    const int b = blockIdx.x;
    const int o = threadIdx.x;
    if (o >= dol) return;
    const int*   si  = selIdx + b * 256;
    const float* sv  = selVal + b * 256;
    const int    cnt = selCnt[b];
    float acc = 0.0f;
    for (int j = 0; j < cnt; ++j) acc += sv[j] * bp[si[j] * dol + o];
    cb[b * dol + o] = acc;
}

// ---------------------------------------------------------------------------
// Per-sample combined weights, f16, row-major [o][i] (flat q = o*di + i):
// cwH[b][q] = sum_j val_j * Wp[idx_j*do*di + q]
// ---------------------------------------------------------------------------
__global__ __launch_bounds__(256) void cw_combine_kernel(
    const float* __restrict__ Wp, const int* __restrict__ selIdx,
    const float* __restrict__ selVal, const int* __restrict__ selCnt,
    _Float16* __restrict__ cwH, int dodi)
{
    const int b = blockIdx.y;
    const int q = blockIdx.x * 256 + threadIdx.x;
    if (q >= dodi) return;
    const int*   si  = selIdx + b * 256;
    const float* sv  = selVal + b * 256;
    const int    cnt = selCnt[b];
    float acc = 0.0f;
    for (int j = 0; j < cnt; ++j) acc += sv[j] * Wp[(size_t)si[j] * dodi + q];
    cwH[(size_t)b * dodi + q] = (_Float16)acc;
}

// ---------------------------------------------------------------------------
// Layer 0: di=2, do=256 -> trivial VALU kernel, writes sin(30x) as f16
// ---------------------------------------------------------------------------
__global__ __launch_bounds__(256) void layer0_kernel(
    const float* __restrict__ coords, const _Float16* __restrict__ cw,
    const float* __restrict__ cb, _Float16* __restrict__ xOut)
{
    const size_t idx = (size_t)blockIdx.x * 256 + threadIdx.x;   // B*P*256 elems
    const int b   = (int)(idx >> 22);
    const int rem = (int)(idx & ((1u << 22) - 1));
    const int p   = rem >> 8;
    const int o   = rem & 255;
    const float c0 = coords[((size_t)b * NP + p) * 2 + 0];
    const float c1 = coords[((size_t)b * NP + p) * 2 + 1];
    const float w0 = (float)cw[b * 512 + o * 2 + 0];
    const float w1 = (float)cw[b * 512 + o * 2 + 1];
    const float v  = c0 * w0 + c1 * w1 + cb[b * NH + o];
    xOut[idx] = (_Float16)sinf(30.0f * v);
}

// ---------------------------------------------------------------------------
// Layers 1..3: Y = sin(30*(X @ cw^T + cb)) via v_wmma_f32_16x16x32_f16.
// 8 waves/block; each wave owns a 32x32 output block (4 WMMA tiles),
// block covers 128 rows x 64 cols. K-loop over 256 in steps of 32.
// A frag (16x32 f16): low lanes K0-7/16-23, high lanes K8-15/24-31 -> two b128.
// B frag (32x16 f16): lane = column; low lanes K0-7/16-23, high lanes K8-15/24-31;
// cwH row-major [o][i] makes each half two contiguous 16B reads.
// ---------------------------------------------------------------------------
__global__ __launch_bounds__(256) void moe_gemm_sin_kernel(
    const _Float16* __restrict__ xIn,   // [NB][NP][NH]
    const _Float16* __restrict__ cwH,   // [NB][NH][NH] row-major [o][i]
    const float*    __restrict__ cb,    // [NB][NH]
    _Float16*       __restrict__ xOut)  // [NB][NP][NH]
{
    const int b       = blockIdx.z;
    const int tid     = threadIdx.x;
    const int wave    = tid >> 5;
    const int lane    = tid & 31;
    const int laneLow = lane & 15;
    const int hi      = lane >> 4;
    const int waveM   = wave & 3;
    const int waveN   = wave >> 2;
    const int r0      = blockIdx.x * 128 + waveM * 32;
    const int n0      = blockIdx.y * 64  + waveN * 32;

    const _Float16* xb = xIn + ((size_t)b * NP + r0) * NH;
    const _Float16* wb = cwH + (size_t)b * NH * NH;

    v8f acc[2][2];
#pragma unroll
    for (int m = 0; m < 2; ++m)
#pragma unroll
        for (int n = 0; n < 2; ++n)
#pragma unroll
            for (int i = 0; i < 8; ++i) acc[m][n][i] = 0.0f;

    for (int kk = 0; kk < NH; kk += 32) {
        v16h a[2], bf[2];
#pragma unroll
        for (int m = 0; m < 2; ++m) {
            const _Float16* p = xb + (size_t)(m * 16 + laneLow) * NH + kk + hi * 8;
            v8h lo = *(const v8h*)(p);
            v8h hh = *(const v8h*)(p + 16);
            a[m] = __builtin_shufflevector(lo, hh, 0,1,2,3,4,5,6,7,8,9,10,11,12,13,14,15);
            __builtin_prefetch((const void*)(p + 32), 0, 0);  // next K step (global_prefetch_b8)
        }
#pragma unroll
        for (int n = 0; n < 2; ++n) {
            const _Float16* p = wb + (size_t)(n0 + n * 16 + laneLow) * NH + kk + hi * 8;
            v8h lo = *(const v8h*)(p);
            v8h hh = *(const v8h*)(p + 16);
            bf[n] = __builtin_shufflevector(lo, hh, 0,1,2,3,4,5,6,7,8,9,10,11,12,13,14,15);
        }
#pragma unroll
        for (int m = 0; m < 2; ++m)
#pragma unroll
            for (int n = 0; n < 2; ++n)
                acc[m][n] = __builtin_amdgcn_wmma_f32_16x16x32_f16(
                    false, a[m], false, bf[n], (short)0, acc[m][n], false, false);
    }

    // Epilogue: C layout -> lanes 0-15: M=r, lanes 16-31: M=r+8, N=laneLow
#pragma unroll
    for (int m = 0; m < 2; ++m) {
#pragma unroll
        for (int n = 0; n < 2; ++n) {
            const int   col  = n0 + n * 16 + laneLow;
            const float bias = cb[b * NH + col];
#pragma unroll
            for (int r = 0; r < 8; ++r) {
                const int row = r0 + m * 16 + r + hi * 8;
                float v = acc[m][n][r] + bias;
                xOut[((size_t)b * NP + row) * NH + col] = (_Float16)sinf(30.0f * v);
            }
        }
    }
}

// ---------------------------------------------------------------------------
// Layer 4: do=3, no sine, f32 output to d_out. One thread per (b,p).
// ---------------------------------------------------------------------------
__global__ __launch_bounds__(256) void layer4_kernel(
    const _Float16* __restrict__ xIn, const _Float16* __restrict__ cw,
    const float* __restrict__ cb, float* __restrict__ out)
{
    const int idx = blockIdx.x * 256 + threadIdx.x;  // B*P threads
    const int b   = idx >> 14;
    const _Float16* xr = xIn + (size_t)idx * NH;
    const _Float16* w  = cw + (size_t)b * 768;       // [3][256]
    float a0 = cb[b * 3 + 0], a1 = cb[b * 3 + 1], a2 = cb[b * 3 + 2];
#pragma unroll 8
    for (int i = 0; i < NH; ++i) {
        float x = (float)xr[i];
        a0 += x * (float)w[i];
        a1 += x * (float)w[NH + i];
        a2 += x * (float)w[2 * NH + i];
    }
    out[(size_t)idx * 3 + 0] = a0;
    out[(size_t)idx * 3 + 1] = a1;
    out[(size_t)idx * 3 + 2] = a2;
}

// ---------------------------------------------------------------------------
extern "C" void kernel_launch(void* const* d_in, const int* in_sizes, int n_in,
                              void* d_out, int out_size, void* d_ws, size_t ws_size,
                              hipStream_t stream) {
    const float* latents = (const float*)d_in[0];
    const float* coords  = (const float*)d_in[1];
    const float *Wp[5], *bp[5], *Wg[5], *bg[5];
    for (int l = 0; l < 5; ++l) {
        Wp[l] = (const float*)d_in[2 + 4 * l];
        bp[l] = (const float*)d_in[3 + 4 * l];
        Wg[l] = (const float*)d_in[4 + 4 * l];
        bg[l] = (const float*)d_in[5 + 4 * l];
    }

    // workspace layout
    uint8_t* ws = (uint8_t*)d_ws;
    size_t cur = 0;
    auto alloc = [&](size_t bytes) -> size_t {
        size_t o = cur;
        cur = (cur + bytes + 255) & ~(size_t)255;
        return o;
    };
    int*      selIdx = (int*)(ws + alloc(5 * NB * 256 * sizeof(int)));
    float*    selVal = (float*)(ws + alloc(5 * NB * 256 * sizeof(float)));
    int*      selCnt = (int*)(ws + alloc(5 * NB * sizeof(int)));
    float*    cbBuf  = (float*)(ws + alloc(NB * NH * sizeof(float)));
    _Float16* cwH    = (_Float16*)(ws + alloc((size_t)NB * NH * NH * sizeof(_Float16)));
    _Float16* xA     = (_Float16*)(ws + alloc((size_t)NB * NP * NH * sizeof(_Float16)));
    _Float16* xB     = (_Float16*)(ws + alloc((size_t)NB * NP * NH * sizeof(_Float16)));

    // 1) gating + top-k for all 5 layers
    gate_topk_kernel<<<dim3(NB, 5), 256, 0, stream>>>(
        latents, Wg[0], Wg[1], Wg[2], Wg[3], Wg[4],
        bg[0], bg[1], bg[2], bg[3], bg[4], selIdx, selVal, selCnt);

    const int DODI[5] = {512, 65536, 65536, 65536, 768};
    const int DOL[5]  = {256, 256, 256, 256, 3};

    // 2) layer 0
    {
        const int l = 0, base = l * NB * 256;
        cb_combine_kernel<<<NB, 256, 0, stream>>>(bp[l], selIdx + base, selVal + base,
                                                  selCnt + l * NB, cbBuf, DOL[l]);
        cw_combine_kernel<<<dim3((DODI[l] + 255) / 256, NB), 256, 0, stream>>>(
            Wp[l], selIdx + base, selVal + base, selCnt + l * NB, cwH, DODI[l]);
        layer0_kernel<<<(NB * NP * NH) / 256, 256, 0, stream>>>(coords, cwH, cbBuf, xA);
    }

    // 3) layers 1..3 (WMMA): xA -> xB -> xA -> xB
    _Float16* xin = xA;
    _Float16* xout = xB;
    for (int l = 1; l <= 3; ++l) {
        const int base = l * NB * 256;
        cb_combine_kernel<<<NB, 256, 0, stream>>>(bp[l], selIdx + base, selVal + base,
                                                  selCnt + l * NB, cbBuf, DOL[l]);
        cw_combine_kernel<<<dim3((DODI[l] + 255) / 256, NB), 256, 0, stream>>>(
            Wp[l], selIdx + base, selVal + base, selCnt + l * NB, cwH, DODI[l]);
        moe_gemm_sin_kernel<<<dim3(NP / 128, NH / 64, NB), 256, 0, stream>>>(
            xin, cwH, cbBuf, xout);
        _Float16* t = xin; xin = xout; xout = t;
    }

    // 4) layer 4 -> d_out (f32). After the swap loop, xin == xB (holds layer-3 out)
    {
        const int l = 4, base = l * NB * 256;
        cb_combine_kernel<<<NB, 256, 0, stream>>>(bp[l], selIdx + base, selVal + base,
                                                  selCnt + l * NB, cbBuf, DOL[l]);
        cw_combine_kernel<<<dim3((DODI[l] + 255) / 256, NB), 256, 0, stream>>>(
            Wp[l], selIdx + base, selVal + base, selCnt + l * NB, cwH, DODI[l]);
        layer4_kernel<<<(NB * NP) / 256, 256, 0, stream>>>(xin, cwH, cbBuf, (float*)d_out);
    }
}
